// PatcherUnfold_6030134084162
// MI455X (gfx1250) — compile-verified, matrix operand order checked by reading, not run on hardware
//
#include <hip/hip_runtime.h>

typedef __attribute__((ext_vector_type(16))) _Float16 v16h;
typedef __attribute__((ext_vector_type(8)))  _Float16 v8h;
typedef __attribute__((ext_vector_type(8)))  float    v8f;

#define PATCH   16
#define GW      24      // patches per image side (384/16)
#define GH      24
#define IMG     384
#define CIN     3
#define KDIM    768     // 3*16*16
#define NDIM    768     // embed dim
#define MDIM    36864   // 64 images * 576 patches
#define BM      256
#define BN      128
#define BK      32
#define NK      (KDIM / BK)   // 24
#define LDSPAD  40      // LDS row stride in halves (80 B, 16B-aligned, bank-skewed)

struct Stage {
    float4 a[8];   // 32 floats: one full A row of this K-block (two image rows)
    float4 b[4];   // 16 floats: half a W row of this K-block
};

__global__ __launch_bounds__(256) void patch_embed_wmma_f16(
    const float* __restrict__ x,      // [64,3,384,384]
    const float* __restrict__ Wm,     // [768,768] (row e, col k)
    const float* __restrict__ bias,   // [768]
    float* __restrict__ out)          // [36864,768]
{
    // ping-pong staged tiles: A 256x32, B 128x32 (f16, padded rows)
    __shared__ __align__(16) _Float16 As[2][BM][LDSPAD];
    __shared__ __align__(16) _Float16 Bs[2][BN][LDSPAD];

    const int tid   = threadIdx.x;
    const int tileN = blockIdx.x % (NDIM / BN);   // varies fastest -> x reuse in L2
    const int tileM = blockIdx.x / (NDIM / BN);

    const int wave   = tid >> 5;
    const int lane   = tid & 31;
    const int wave_m = wave >> 1;      // 0..3, 64 rows each
    const int wave_n = wave & 1;       // 0..1, 64 cols each
    const int rsub   = lane & 15;
    const int selK   = (lane < 16) ? 0 : 8;   // per-half-wave K sub-offset (ISA layout)

    // ---- A staging: thread t owns full row t of the 256x32 tile ----
    const int m  = tileM * BM + tid;
    const int bi = m / (GH * GW);
    const int l  = m % (GH * GW);
    const int gy = l / GW;
    const int gx = l % GW;

    // ---- B staging: thread t owns (row t/2, 16-K half t&1) of the 128x32 tile ----
    const int brow  = tid >> 1;
    const int bhalf = tid & 1;
    const int e     = tileN * BN + brow;

    auto fetch = [&](int kbS, Stage& s) {
        const int k0  = kbS * BK;
        const int c   = k0 >> 8;              // 256 k-values per input channel
        const int kh0 = (k0 & 255) >> 4;      // starting kernel row (k0 % 16 == 0)
        const int y0  = gy * PATCH + kh0;
        const float* abase = x + (((size_t)bi * CIN + c) * IMG + y0) * IMG + (size_t)gx * PATCH;
        const float4* pa0 = (const float4*)abase;
        const float4* pa1 = (const float4*)(abase + IMG);
        const float4* pb  = (const float4*)(Wm + (size_t)e * KDIM + k0 + bhalf * PATCH);
#pragma unroll
        for (int q = 0; q < 4; ++q) s.a[q]     = pa0[q];
#pragma unroll
        for (int q = 0; q < 4; ++q) s.a[4 + q] = pa1[q];
#pragma unroll
        for (int q = 0; q < 4; ++q) s.b[q]     = pb[q];
        // prefetch the following K-block's A source (emits global_prefetch_b8)
        if (kbS + 1 < NK) {
            const int k0n  = k0 + BK;
            const int cn   = k0n >> 8;
            const int kh0n = (k0n & 255) >> 4;
            __builtin_prefetch(x + (((size_t)bi * CIN + cn) * IMG + gy * PATCH + kh0n) * IMG
                                 + (size_t)gx * PATCH, 0, 1);
        }
    };

    auto commit = [&](const Stage& s, int bufS) {
#pragma unroll
        for (int q = 0; q < 4; ++q) {
            v8h h = { (_Float16)s.a[2*q].x, (_Float16)s.a[2*q].y,
                      (_Float16)s.a[2*q].z, (_Float16)s.a[2*q].w,
                      (_Float16)s.a[2*q+1].x, (_Float16)s.a[2*q+1].y,
                      (_Float16)s.a[2*q+1].z, (_Float16)s.a[2*q+1].w };
            *(v8h*)&As[bufS][tid][q * 8] = h;
        }
#pragma unroll
        for (int q = 0; q < 2; ++q) {
            v8h h = { (_Float16)s.b[2*q].x, (_Float16)s.b[2*q].y,
                      (_Float16)s.b[2*q].z, (_Float16)s.b[2*q].w,
                      (_Float16)s.b[2*q+1].x, (_Float16)s.b[2*q+1].y,
                      (_Float16)s.b[2*q+1].z, (_Float16)s.b[2*q+1].w };
            *(v8h*)&Bs[bufS][brow][bhalf * 16 + q * 8] = h;
        }
    };

    v8f acc[4][4];
#pragma unroll
    for (int i = 0; i < 4; ++i)
#pragma unroll
        for (int j = 0; j < 4; ++j)
            acc[i][j] = (v8f){};

    // ---- prologue: stage K-block 0 into buffer 0 ----
    {
        Stage s0;
        fetch(0, s0);
        commit(s0, 0);
    }
    __syncthreads();

    for (int kb = 0; kb < NK; ++kb) {
        const int cur  = kb & 1;
        const bool more = (kb + 1 < NK);

        // issue next K-block's global loads before compute (latency hidden by WMMA)
        Stage nst;
        if (more) fetch(kb + 1, nst);

        // ---- fragments per ISA 16-bit A/B layout ----
        v16h afrag[4], bfrag[4];
#pragma unroll
        for (int i = 0; i < 4; ++i) {
            const int r = wave_m * 64 + i * 16 + rsub;
            v8h lo = *(const v8h*)&As[cur][r][selK];        // K 0..7  (+8 upper half-wave)
            v8h hi = *(const v8h*)&As[cur][r][16 + selK];   // K 16..23 (+8)
            afrag[i] = __builtin_shufflevector(lo, hi, 0,1,2,3,4,5,6,7,8,9,10,11,12,13,14,15);
        }
#pragma unroll
        for (int j = 0; j < 4; ++j) {
            const int r = wave_n * 64 + j * 16 + rsub;
            v8h lo = *(const v8h*)&Bs[cur][r][selK];
            v8h hi = *(const v8h*)&Bs[cur][r][16 + selK];
            bfrag[j] = __builtin_shufflevector(lo, hi, 0,1,2,3,4,5,6,7,8,9,10,11,12,13,14,15);
        }

        // ---- 16 WMMAs per K step (64x64 per wave), f32 accumulate ----
#pragma unroll
        for (int i = 0; i < 4; ++i)
#pragma unroll
            for (int j = 0; j < 4; ++j)
                acc[i][j] = __builtin_amdgcn_wmma_f32_16x16x32_f16(
                    false, afrag[i], false, bfrag[j],
                    (short)0, acc[i][j], false, false);

        // convert + store next block into the other buffer, then one barrier
        if (more) commit(nst, cur ^ 1);
        __syncthreads();
    }

    // ---- epilogue: bias add + store (f32 C/D: lanes 0-15 rows v, 16-31 rows v+8) ----
#pragma unroll
    for (int j = 0; j < 4; ++j) {
        const int n = tileN * BN + wave_n * 64 + j * 16 + rsub;
        const float bv = bias[n];
#pragma unroll
        for (int i = 0; i < 4; ++i) {
            const int mbase = tileM * BM + wave_m * 64 + i * 16 + ((lane & 16) ? 8 : 0);
#pragma unroll
            for (int v = 0; v < 8; ++v)
                out[(size_t)(mbase + v) * NDIM + n] = acc[i][j][v] + bv;
        }
    }
}

extern "C" void kernel_launch(void* const* d_in, const int* in_sizes, int n_in,
                              void* d_out, int out_size, void* d_ws, size_t ws_size,
                              hipStream_t stream) {
    const float* x    = (const float*)d_in[0];   // [64,3,384,384]
    const float* Wm   = (const float*)d_in[1];   // [768,768]
    const float* bias = (const float*)d_in[2];   // [768]
    float* out        = (float*)d_out;           // [64,576,768]

    const int grid = (MDIM / BM) * (NDIM / BN);  // 144 * 6 = 864 workgroups
    patch_embed_wmma_f16<<<dim3(grid), dim3(256), 0, stream>>>(x, Wm, bias, out);
}
